// DPSF_21311627722836
// MI455X (gfx1250) — matrix operand order repaired and checked by reading, not compiled, not run on hardware
//
#include <hip/hip_runtime.h>
#include <hip/hip_bf16.h>
#include <math.h>

// ---------------- problem constants ----------------
#define B_   4
#define L_   2048
#define DM   768
#define DSTATE 64
#define DCONV 4
#define DIN  1536            // EXPAND * DM
#define DTR  48              // ceil(DM/16)
#define BL   (B_ * L_)       // 8192 rows in every GEMM
#define XDBL_N (DTR + 2 * DSTATE)   // 176
#define XDBL_NPAD 192        // 176 padded to 64-multiple
#define DTR_PAD 64           // 48 padded to 64

typedef __attribute__((ext_vector_type(16))) __bf16 v16bf;
typedef __attribute__((ext_vector_type(8)))  __bf16 v8bf;
typedef __attribute__((ext_vector_type(8)))  float  v8f;

// ---------------------------------------------------------------------------
// fp32 -> bf16 convert with optional 2-D zero-padding.
//   src: srcRows x srcCols (row stride srcStride, fp32)
//   dst: dstRows x dstCols (contiguous, bf16); OOB of src -> 0
// ---------------------------------------------------------------------------
__global__ __launch_bounds__(256)
void cvt_pad_bf16(const float* __restrict__ src, __bf16* __restrict__ dst,
                  int srcRows, int srcCols, int srcStride,
                  int dstRows, int dstCols)
{
    const size_t idx = (size_t)blockIdx.x * blockDim.x + threadIdx.x;
    if (idx >= (size_t)dstRows * dstCols) return;
    const int r = (int)(idx / dstCols);
    const int c = (int)(idx % dstCols);
    float v = (r < srcRows && c < srcCols) ? src[(size_t)r * srcStride + c] : 0.f;
    dst[idx] = (__bf16)v;
}

// ---------------------------------------------------------------------------
// WMMA GEMM:  C[M x N] = A[M x K] * W[N' x K]^T     A,W bf16; C fp32
// Requirements: M%128==0, K%64==0, N' (padded W rows) %64==0, lda%8==0.
// Real N used only for the store guard. mode 1: C = softplus(acc + bias[n]).
// Block tile 128x64, K-step 64, software-pipelined global->LDS staging,
// 8 x v_wmma_f32_16x16x32_bf16 per K-step per wave.
// ---------------------------------------------------------------------------
#define LDSA 72   // 64 + 8 pad (bf16 elems); row = 144 B (16B multiple)
#define LDSB 72

__global__ __launch_bounds__(256)
void gemm_bf16_wmma(const __bf16* __restrict__ A,
                    const __bf16* __restrict__ W,
                    float* __restrict__ C,
                    int N, int K, int lda, int ldc,
                    const float* __restrict__ bias, int mode)
{
    __shared__ __bf16 sA[128 * LDSA];
    __shared__ __bf16 sB[64 * LDSB];

    const int tid    = threadIdx.x;
    const int lane   = tid & 31;
    const int wave   = tid >> 5;          // 0..7
    const int wm     = wave & 3;          // 4 waves along M (32 rows each)
    const int wn     = wave >> 2;         // 2 waves along N (32 cols each)
    const int blockM = blockIdx.x * 128;
    const int blockN = blockIdx.y * 64;

    // staging geometry: chunks of 8 bf16 (16 B)
    // A tile 128x64 -> 1024 chunks (4/thread); B tile 64x64 -> 512 (2/thread)
    uint4 aReg[4], bReg[2];

    auto loadTiles = [&](int k0) {
        #pragma unroll
        for (int i = 0; i < 4; ++i) {
            const int ch = tid + 256 * i;
            const int r = ch >> 3, c = (ch & 7) << 3;
            aReg[i] = *(const uint4*)(A + (size_t)(blockM + r) * lda + k0 + c);
        }
        #pragma unroll
        for (int i = 0; i < 2; ++i) {
            const int ch = tid + 256 * i;
            const int r = ch >> 3, c = (ch & 7) << 3;
            bReg[i] = *(const uint4*)(W + (size_t)(blockN + r) * K + k0 + c);
        }
    };
    auto storeTiles = [&]() {
        #pragma unroll
        for (int i = 0; i < 4; ++i) {
            const int ch = tid + 256 * i;
            const int r = ch >> 3, c = (ch & 7) << 3;
            *(uint4*)(sA + r * LDSA + c) = aReg[i];
        }
        #pragma unroll
        for (int i = 0; i < 2; ++i) {
            const int ch = tid + 256 * i;
            const int r = ch >> 3, c = (ch & 7) << 3;
            *(uint4*)(sB + r * LDSB + c) = bReg[i];
        }
    };

    v8f acc[2][2];
    #pragma unroll
    for (int i = 0; i < 2; ++i)
        #pragma unroll
        for (int j = 0; j < 2; ++j)
            acc[i][j] = (v8f){0.f,0.f,0.f,0.f,0.f,0.f,0.f,0.f};

    const int ksteps = K >> 6;
    loadTiles(0);
    storeTiles();
    __syncthreads();

    const int arl = lane & 15;                 // row/col within 16
    const int kbA = (lane < 16) ? 0 : 8;       // A frag K-base (16-bit layout)
    const int kbB = (lane < 16) ? 0 : 8;       // B frag K-base

    for (int ks = 0; ks < ksteps; ++ks) {
        const bool more = (ks + 1) < ksteps;
        if (more) {
            loadTiles((ks + 1) << 6);
            if (ks + 2 < ksteps)               // hint L2 one more tile ahead
                __builtin_prefetch(A + (size_t)(blockM + (tid >> 1)) * lda
                                     + ((ks + 2) << 6), 0, 1);
        }

        #pragma unroll
        for (int kh = 0; kh < 2; ++kh) {       // two 32-wide K halves
            v16bf af[2], bfr[2];
            #pragma unroll
            for (int ti = 0; ti < 2; ++ti) {
                const __bf16* ap = sA + (wm*32 + ti*16 + arl) * LDSA + kh*32;
                v8bf lo = *(const v8bf*)(ap + kbA);
                v8bf hi = *(const v8bf*)(ap + kbA + 16);
                #pragma unroll
                for (int e = 0; e < 8; ++e) { af[ti][e] = lo[e]; af[ti][8+e] = hi[e]; }
            }
            #pragma unroll
            for (int tj = 0; tj < 2; ++tj) {
                const __bf16* bp = sB + (wn*32 + tj*16 + arl) * LDSB + kh*32 + kbB;
                v8bf lo = *(const v8bf*)(bp);
                v8bf hi = *(const v8bf*)(bp + 16);
                #pragma unroll
                for (int e = 0; e < 8; ++e) { bfr[tj][e] = lo[e]; bfr[tj][8+e] = hi[e]; }
            }
            #pragma unroll
            for (int ti = 0; ti < 2; ++ti)
                #pragma unroll
                for (int tj = 0; tj < 2; ++tj)
                    acc[ti][tj] = __builtin_amdgcn_wmma_f32_16x16x32_bf16(
                        false, af[ti], false, bfr[tj],
                        (short)0, acc[ti][tj], false, false);
        }

        __syncthreads();
        if (more) storeTiles();
        __syncthreads();
    }

    // epilogue: C/D layout -> VGPR r: M=base+r; lanes 0-15 N=lane (M+0),
    // lanes 16-31 N=lane-16 (M+8)
    const int ln = lane & 15;
    #pragma unroll
    for (int ti = 0; ti < 2; ++ti) {
        const int mbase = blockM + wm*32 + ti*16 + ((lane < 16) ? 0 : 8);
        #pragma unroll
        for (int tj = 0; tj < 2; ++tj) {
            const int gn = blockN + wn*32 + tj*16 + ln;
            if (gn >= N) continue;
            const float bval = (mode == 1) ? bias[gn] : 0.f;
            #pragma unroll
            for (int r = 0; r < 8; ++r) {
                float v = acc[ti][tj][r];
                if (mode == 1) {
                    v += bval;
                    v = (v > 20.f) ? v : log1pf(__expf(v));
                }
                C[(size_t)(mbase + r) * ldc + gn] = v;
            }
        }
    }
}

// ---------------------------------------------------------------------------
// Fused causal depthwise conv (DCONV=4) + SiLU, both branches.
// x2 fp32 (scan input), x2b bf16 (x-proj GEMM A), sr fp32 (gate).
// ---------------------------------------------------------------------------
__device__ __forceinline__ float silu_f(float x) {
    return x * (1.f / (1.f + __expf(-x)));
}

__global__ __launch_bounds__(256)
void conv_silu_kernel(const float* __restrict__ xr,
                      const float* __restrict__ conv_w,
                      const float* __restrict__ conv_b,
                      float* __restrict__ x2,
                      __bf16* __restrict__ x2b,
                      float* __restrict__ sr)
{
    const size_t idx = (size_t)blockIdx.x * blockDim.x + threadIdx.x;
    if (idx >= (size_t)BL * DIN) return;
    const int d  = (int)(idx % DIN);
    const int bl = (int)(idx / DIN);
    const int l  = bl % L_;
    const int b  = bl / L_;

    const float w0 = conv_w[d * 4 + 0];
    const float w1 = conv_w[d * 4 + 1];
    const float w2 = conv_w[d * 4 + 2];
    const float w3 = conv_w[d * 4 + 3];
    const float bb = conv_b[d];

    float a0 = bb, a1 = bb;
    #pragma unroll
    for (int j = 0; j < DCONV; ++j) {
        const int lt = l - (DCONV - 1) + j;
        if (lt < 0) continue;
        const float w = (j == 0) ? w0 : (j == 1) ? w1 : (j == 2) ? w2 : w3;
        const size_t base = ((size_t)(b * L_ + lt)) * (2 * DIN);
        a0 += w * xr[base + d];
        a1 += w * xr[base + DIN + d];
    }
    const float s0 = silu_f(a0);
    x2[idx]  = s0;
    x2b[idx] = (__bf16)s0;
    sr[idx]  = silu_f(a1);
}

// ---------------------------------------------------------------------------
// Selective scan. grid = (DIN/64, B), block = 64 threads (2 waves).
// One channel per thread; A-row + 64-state in VGPRs; B_t/C_t via LDS.
// Writes gated output directly as bf16 for the out-proj GEMM.
// ---------------------------------------------------------------------------
__global__ __launch_bounds__(64)
void scan_kernel(const float* __restrict__ x2,
                 const float* __restrict__ x_dbl,
                 const float* __restrict__ delta,
                 const float* __restrict__ A_log,
                 const float* __restrict__ D_param,
                 const float* __restrict__ sr,
                 __bf16* __restrict__ y_gated)
{
    const int d = blockIdx.x * 64 + threadIdx.x;
    const int b = blockIdx.y;
    const int tid = threadIdx.x;

    __shared__ float Bs[DSTATE];
    __shared__ float Cs[DSTATE];

    float Arow[DSTATE], st[DSTATE];
    #pragma unroll
    for (int n = 0; n < DSTATE; ++n) {
        Arow[n] = -__expf(A_log[(size_t)d * DSTATE + n]);
        st[n]   = 0.f;
    }
    const float Dp = D_param[d];

    for (int t = 0; t < L_; ++t) {
        const size_t row = (size_t)(b * L_ + t);
        Bs[tid] = x_dbl[row * XDBL_N + DTR + tid];
        Cs[tid] = x_dbl[row * XDBL_N + DTR + DSTATE + tid];
        __syncthreads();

        const size_t didx = row * DIN + d;
        const float dt = delta[didx];
        const float u  = x2[didx];
        const float du = dt * u;
        float y = 0.f;
        #pragma unroll
        for (int n = 0; n < DSTATE; ++n) {
            st[n] = __expf(dt * Arow[n]) * st[n] + du * Bs[n];
            y += st[n] * Cs[n];
        }
        y += u * Dp;
        y_gated[didx] = (__bf16)(y * sr[didx]);
        __syncthreads();
    }
}

// ---------------------------------------------------------------------------
// Host-side orchestration (graph-capture safe)
// ---------------------------------------------------------------------------
static inline unsigned cdiv_u(size_t a, unsigned b) { return (unsigned)((a + b - 1) / b); }

extern "C" void kernel_launch(void* const* d_in, const int* in_sizes, int n_in,
                              void* d_out, int out_size, void* d_ws, size_t ws_size,
                              hipStream_t stream)
{
    const float* x       = (const float*)d_in[0];   // (B,L,DM)
    const float* W_in    = (const float*)d_in[1];   // (3072, 768)
    const float* conv_w  = (const float*)d_in[2];   // (1536,1,4)
    const float* conv_b  = (const float*)d_in[3];   // (1536)
    const float* W_xproj = (const float*)d_in[4];   // (176, 1536)
    const float* W_dt    = (const float*)d_in[5];   // (1536, 48)
    const float* b_dt    = (const float*)d_in[6];   // (1536)
    const float* A_log   = (const float*)d_in[7];   // (1536, 64)
    const float* D_param = (const float*)d_in[8];   // (1536)
    const float* W_out   = (const float*)d_in[9];   // (768, 1536)
    float* out = (float*)d_out;                     // (B,L,DM) fp32

    // ---- workspace carve-up ----
    // fp32 region
    float* xr    = (float*)d_ws;                        // BL x 3072
    float* x2    = xr   + (size_t)BL * (2 * DIN);       // BL x 1536
    float* sr    = x2   + (size_t)BL * DIN;             // BL x 1536
    float* xdbl  = sr   + (size_t)BL * DIN;             // BL x 176
    float* dlt   = xdbl + (size_t)BL * XDBL_N;          // BL x 1536
    // bf16 region (16B-aligned: fp32 region is a multiple of 16 bytes)
    __bf16* xb    = (__bf16*)(dlt + (size_t)BL * DIN);  // BL x 768
    __bf16* Winb  = xb    + (size_t)BL * DM;            // 3072 x 768
    __bf16* Wxpb  = Winb  + (size_t)(2 * DIN) * DM;     // 192 x 1536 (padded)
    __bf16* Wdtb  = Wxpb  + (size_t)XDBL_NPAD * DIN;    // 1536 x 64  (padded)
    __bf16* Woutb = Wdtb  + (size_t)DIN * DTR_PAD;      // 768 x 1536
    __bf16* x2b   = Woutb + (size_t)DM * DIN;           // BL x 1536
    __bf16* dtb   = x2b   + (size_t)BL * DIN;           // BL x 64 (padded dt-part)
    __bf16* ygb   = dtb   + (size_t)BL * DTR_PAD;       // BL x 1536

    const dim3 blk(256);

    // ---- operand conversions (bf16, with padding where needed) ----
    cvt_pad_bf16<<<cdiv_u((size_t)BL * DM, 256), blk, 0, stream>>>(
        x, xb, BL, DM, DM, BL, DM);
    cvt_pad_bf16<<<cdiv_u((size_t)(2 * DIN) * DM, 256), blk, 0, stream>>>(
        W_in, Winb, 2 * DIN, DM, DM, 2 * DIN, DM);
    cvt_pad_bf16<<<cdiv_u((size_t)XDBL_NPAD * DIN, 256), blk, 0, stream>>>(
        W_xproj, Wxpb, XDBL_N, DIN, DIN, XDBL_NPAD, DIN);
    cvt_pad_bf16<<<cdiv_u((size_t)DIN * DTR_PAD, 256), blk, 0, stream>>>(
        W_dt, Wdtb, DIN, DTR, DTR, DIN, DTR_PAD);
    cvt_pad_bf16<<<cdiv_u((size_t)DM * DIN, 256), blk, 0, stream>>>(
        W_out, Woutb, DM, DIN, DIN, DM, DIN);

    // 1) in-projection: (8192x768)*(3072x768)^T -> xr fp32
    gemm_bf16_wmma<<<dim3(BL / 128, (2 * DIN) / 64), blk, 0, stream>>>(
        xb, Winb, xr, 2 * DIN, DM, DM, 2 * DIN, nullptr, 0);

    // 2) causal depthwise conv + SiLU (both branches)
    conv_silu_kernel<<<cdiv_u((size_t)BL * DIN, 256), blk, 0, stream>>>(
        xr, conv_w, conv_b, x2, x2b, sr);

    // 3) x-projection: (8192x1536)*(192x1536)^T -> xdbl (N guard 176)
    gemm_bf16_wmma<<<dim3(BL / 128, XDBL_NPAD / 64), blk, 0, stream>>>(
        x2b, Wxpb, xdbl, XDBL_N, DIN, DIN, XDBL_N, nullptr, 0);

    // 3b) pack delta-part of xdbl into K=64 zero-padded bf16
    cvt_pad_bf16<<<cdiv_u((size_t)BL * DTR_PAD, 256), blk, 0, stream>>>(
        xdbl, dtb, BL, DTR, XDBL_N, BL, DTR_PAD);

    // 4) delta = softplus(dt_part * W_dt^T + b_dt)
    gemm_bf16_wmma<<<dim3(BL / 128, DIN / 64), blk, 0, stream>>>(
        dtb, Wdtb, dlt, DIN, DTR_PAD, DTR_PAD, DIN, b_dt, 1);

    // 5) selective scan + gating -> y_gated bf16
    scan_kernel<<<dim3(DIN / 64, B_), dim3(64), 0, stream>>>(
        x2, xdbl, dlt, A_log, D_param, sr, ygb);

    // 6) out-projection: (8192x1536)*(768x1536)^T -> d_out fp32
    gemm_bf16_wmma<<<dim3(BL / 128, DM / 64), blk, 0, stream>>>(
        ygb, Woutb, out, DM, DIN, DIN, DM, nullptr, 0);

    (void)in_sizes; (void)n_in; (void)out_size; (void)ws_size;
}
